// DecoderWithRNN_17394617549451
// MI455X (gfx1250) — matrix operand order repaired and checked by reading, not compiled, not run on hardware
//
#include <hip/hip_runtime.h>
#include <hip/hip_bf16.h>

typedef unsigned int  u32;
typedef unsigned short ushort_t;
typedef __attribute__((ext_vector_type(16))) __bf16 bf16x16;
typedef __attribute__((ext_vector_type(8)))  float  f32x8;
typedef __attribute__((ext_vector_type(2)))  float  f32x2;

union ABf { bf16x16 v; u32 u[8]; };

#define B_   32
#define L_   52
#define V_   10000
#define H_   512
#define T_   51
#define ENC_ 401408
#define G4_  2048    // 4*H

__device__ __forceinline__ ushort_t f2bf(float f) {
    u32 u = __float_as_uint(f);
    u += 0x7FFFu + ((u >> 16) & 1u);       // round-to-nearest-even
    return (ushort_t)(u >> 16);
}
__device__ __forceinline__ float bf2f(ushort_t h) {
    return __uint_as_float((u32)h << 16);
}
__device__ __forceinline__ u32 pack2(float lo, float hi) {
    return (u32)f2bf(lo) | ((u32)f2bf(hi) << 16);
}
// A-matrix 16x32 bf16 layout: lane<16 holds K {0..7,16..23}, lane>=16 holds {8..15,24..31}
__device__ __forceinline__ int a_koff(int v, int hi) {
    return 2 * v + (v >= 4 ? 8 : 0) + hi * 8;
}
__device__ __forceinline__ float sigf(float x) {
    return 1.0f / (1.0f + __expf(-x));
}
__device__ __forceinline__ f32x8 wmma_bf16(const ABf& a, const ABf& b, f32x8 c) {
    return __builtin_amdgcn_wmma_f32_16x16x32_bf16(false, a.v, false, b.v,
                                                   (short)0, c, false, false);
}
__device__ __forceinline__ f32x8 wmma_f32(f32x2 a, f32x2 b, f32x8 c) {
    return __builtin_amdgcn_wmma_f32_16x16x4_f32(false, a, false, b,
                                                 (short)0, c, false, false);
}
// --- CDNA5 async global->LDS copy (16B per lane), tracked by ASYNCcnt ---
__device__ __forceinline__ void async_copy16(u32 lds_addr, const void* gaddr) {
    asm volatile("global_load_async_to_lds_b128 %0, %1, off"
                 :: "v"(lds_addr), "v"(gaddr) : "memory");
}
__device__ __forceinline__ void wait_async0() {
    asm volatile("s_wait_asynccnt 0x0" ::: "memory");
}

// ---------------------------------------------------------------- prep: sort
__global__ void prep_kernel(const int* __restrict__ caps_in, const int* __restrict__ cap_len,
                            const float* __restrict__ init_b,
                            int* sort_ind, int* dec_len, int* caps_sorted,
                            float* x0, float* out_tail) {
    __shared__ int s_ind[B_];
    int tid = threadIdx.x;
    if (tid < B_) {
        int li = cap_len[tid];
        int rank = 0;
        for (int j = 0; j < B_; ++j) {
            int lj = cap_len[j];
            if (lj > li || (lj == li && j < tid)) rank++;
        }
        s_ind[rank] = tid;   // stable descending argsort
    }
    __syncthreads();
    if (tid < B_) {
        int src = s_ind[tid];
        sort_ind[tid] = src;
        dec_len[tid]  = cap_len[src] - 1;
        out_tail[B_ * L_ + tid]       = (float)(cap_len[src] - 1); // dec_len out
        out_tail[B_ * L_ + B_ + tid]  = (float)src;                // sort_ind out
    }
    __syncthreads();
    for (int i = tid; i < B_ * L_; i += blockDim.x) {
        int b = i / L_, l = i % L_;
        int v = caps_in[s_ind[b] * L_ + l];
        caps_sorted[i] = v;
        out_tail[i] = (float)v;                                    // caps out
    }
    for (int i = tid; i < B_ * H_; i += blockDim.x)
        x0[i] = init_b[i & (H_ - 1)];                              // bias init for atomics
}

// -------------------------------------------------- weight fp32 -> bf16 once
__global__ void convert_kernel(const float* __restrict__ w_ih, const float* __restrict__ w_hh,
                               const float* __restrict__ fc_w,
                               const float* __restrict__ b_ih, const float* __restrict__ b_hh,
                               ushort_t* w_ih_b, ushort_t* w_hh_b, ushort_t* fc_w_b,
                               float* b_comb) {
    size_t i = (size_t)blockIdx.x * blockDim.x + threadIdx.x;
    size_t stride = (size_t)gridDim.x * blockDim.x;
    for (size_t k = i; k < (size_t)G4_ * H_; k += stride) {
        w_ih_b[k] = f2bf(w_ih[k]);
        w_hh_b[k] = f2bf(w_hh[k]);
    }
    for (size_t k = i; k < (size_t)V_ * H_; k += stride) fc_w_b[k] = f2bf(fc_w[k]);
    for (size_t k = i; k < (size_t)G4_; k += stride)     b_comb[k] = b_ih[k] + b_hh[k];
}

// ---- x0 = enc[sort] @ init_w.T  (K-split, HBM-bound, full-precision f32 WMMA)
__global__ void initmm_kernel(const float* __restrict__ enc, const float* __restrict__ init_w,
                              const int* __restrict__ sort_ind, float* x0) {
    const int NT = H_ / 16;        // 32 n-tiles
    const int KSEG = 98;           // 98 * 4096 == ENC_
    int wid = (blockIdx.x * blockDim.x + threadIdx.x) >> 5;
    if (wid >= NT * KSEG) return;
    int lane = threadIdx.x & 31, hi = lane >> 4, l15 = lane & 15;
    int nt = wid % NT, ks = wid / NT;
    // f32 WMMA 16x16x4: A lane = row l15, K = 2*hi + {0,1}  -> contiguous float2
    //                   B lane = col l15, K = 2*hi + {0,1}  -> contiguous float2
    const float* arow0 = enc + (size_t)sort_ind[l15] * ENC_ + 2 * hi;
    const float* arow1 = enc + (size_t)sort_ind[16 + l15] * ENC_ + 2 * hi;
    const float* brow  = init_w + (size_t)(nt * 16 + l15) * ENC_ + 2 * hi;
    f32x8 acc0 = {}, acc1 = {};
    int k0 = ks * 4096;
    for (int kc = 0; kc < 4096; kc += 64) {
        __builtin_prefetch(brow + k0 + kc + 2048, 0, 1);   // global_prefetch 2048 floats (8KB) ahead
#pragma unroll
        for (int kk = 0; kk < 64; kk += 4) {
            int k = k0 + kc + kk;
            f32x2 a0 = *(const f32x2*)(arow0 + k);
            f32x2 a1 = *(const f32x2*)(arow1 + k);
            f32x2 bb = *(const f32x2*)(brow  + k);
            acc0 = wmma_f32(a0, bb, acc0);
            acc1 = wmma_f32(a1, bb, acc1);
        }
    }
    int n = nt * 16 + l15;
#pragma unroll
    for (int r = 0; r < 8; ++r) {
        atomicAdd(&x0[(size_t)(r + hi * 8) * H_ + n],        acc0[r]);
        atomicAdd(&x0[(size_t)(16 + r + hi * 8) * H_ + n],   acc1[r]);
    }
}

// ------------------------------------------------ batchnorm over batch axis
__global__ void bn_kernel(const float* __restrict__ x0, const float* __restrict__ gamma,
                          const float* __restrict__ beta, ushort_t* x0bn) {
    int h = threadIdx.x;           // 512 threads
    float mu = 0.f;
    for (int b = 0; b < B_; ++b) mu += x0[b * H_ + h];
    mu *= (1.f / B_);
    float var = 0.f;
    for (int b = 0; b < B_; ++b) { float d = x0[b * H_ + h] - mu; var += d * d; }
    var *= (1.f / B_);
    float sc = rsqrtf(var + 1e-5f) * gamma[h];
    float bt = beta[h];
    for (int b = 0; b < B_; ++b)
        x0bn[b * H_ + h] = f2bf((x0[b * H_ + h] - mu) * sc + bt);
}

// ---- XW[s] = input_s @ w_ih.T + (b_ih+b_hh) in bf16;  s=0 -> BN(x0), s>0 -> emb
__global__ void xw_kernel(const float* __restrict__ emb, const int* __restrict__ caps_sorted,
                          const ushort_t* __restrict__ x0bn, const ushort_t* __restrict__ w_ih_b,
                          const float* __restrict__ b_comb, ushort_t* __restrict__ XW_b) {
    const int MT = (L_ * B_) / 16; // 104
    const int NT = G4_ / 16;       // 128
    int wid = (blockIdx.x * blockDim.x + threadIdx.x) >> 5;
    if (wid >= MT * NT) return;
    int lane = threadIdx.x & 31, hi = lane >> 4, l15 = lane & 15;
    int nt = wid % NT, mt = wid / NT;
    int m = mt * 16 + l15;
    int s = m >> 5, b = m & 31;
    const u32* xrow = (const u32*)(x0bn + b * H_);
    const float* erow = (s > 0) ? (emb + (size_t)caps_sorted[b * L_ + (s - 1)] * H_) : nullptr;
    const u32* brow = (const u32*)(w_ih_b + (size_t)(nt * 16 + l15) * H_);
    f32x8 acc = {};
    for (int k0 = 0; k0 < H_; k0 += 32) {
        ABf a, bb;
#pragma unroll
        for (int v = 0; v < 8; ++v) {
            int ka = k0 + a_koff(v, hi);
            a.u[v]  = (s == 0) ? xrow[ka >> 1] : pack2(erow[ka], erow[ka + 1]);
            bb.u[v] = brow[(k0 + a_koff(v, hi)) >> 1];
        }
        acc = wmma_bf16(a, bb, acc);
    }
    int n = nt * 16 + l15;
#pragma unroll
    for (int r = 0; r < 8; ++r) {
        int mrow = mt * 16 + r + hi * 8;
        XW_b[(size_t)mrow * G4_ + n] = f2bf(acc[r] + b_comb[n]);
    }
}

// -------- persistent single-block LSTM recurrence --------------------------
// h in LDS (bf16), c in VGPRs, XW[it] double-buffered in LDS via
// global_load_async_to_lds_b128 (ASYNCcnt) overlapped with the h@w_hh WMMAs.
__global__ __launch_bounds__(1024) void recur_kernel(const ushort_t* __restrict__ XW_b,
                                                     const ushort_t* __restrict__ w_hh_b,
                                                     const int* __restrict__ dec_len,
                                                     ushort_t* __restrict__ H_all) {
    __shared__ ushort_t hls[B_ * H_];          // 32 KB hidden state
    __shared__ ushort_t xwlds[2][B_ * G4_];    // 2 x 128 KB gate pre-activations
    int tid = threadIdx.x;
    int wid = tid >> 5, lane = tid & 31, hi = lane >> 4, l15 = lane & 15;
    int ng = wid;                              // wave owns columns [ng*16, ng*16+16)
    float c[2][8];
#pragma unroll
    for (int mt = 0; mt < 2; ++mt)
#pragma unroll
        for (int r = 0; r < 8; ++r) c[mt][r] = 0.f;

    // stage XW[0] into buffer 0 (async DMA, 128KB: 1024 threads x 8 x 16B)
    {
        const char* src = (const char*)XW_b;
#pragma unroll
        for (int j = 0; j < 8; ++j) {
            int boff = tid * 16 + j * 16384;
            async_copy16((u32)(size_t)((const char*)&xwlds[0][0] + boff), src + boff);
        }
        wait_async0();
    }
    __syncthreads();

    for (int it = 0; it <= T_; ++it) {    // it=0: init cell (x0), it>=1: scan step t=it-1
        int buf = it & 1;
        // kick off async staging of XW[it+1] into the idle buffer; the DMA
        // overlaps this step's WMMA work and is awaited at end of iteration.
        if (it < T_) {
            const char* src = (const char*)(XW_b + (size_t)(it + 1) * B_ * G4_);
#pragma unroll
            for (int j = 0; j < 8; ++j) {
                int boff = tid * 16 + j * 16384;
                async_copy16((u32)(size_t)((const char*)&xwlds[buf ^ 1][0] + boff),
                             src + boff);
            }
        }
        float hnew[2][8];
#pragma unroll
        for (int mt = 0; mt < 2; ++mt) {
            f32x8 g[4];
            int ncol = ng * 16 + l15;
#pragma unroll
            for (int gi = 0; gi < 4; ++gi)
#pragma unroll
                for (int r = 0; r < 8; ++r)
                    g[gi][r] = bf2f(xwlds[buf][(mt * 16 + r + hi * 8) * G4_ + gi * H_ + ncol]);
            if (it > 0) {                 // h_prev == 0 at it==0: skip matmul
                const u32* bptr[4];
#pragma unroll
                for (int gi = 0; gi < 4; ++gi)
                    bptr[gi] = (const u32*)(w_hh_b + (size_t)(gi * H_ + ng * 16 + l15) * H_);
                int arowbase = (mt * 16 + l15) * H_;
                for (int k0 = 0; k0 < H_; k0 += 32) {
                    ABf a;
#pragma unroll
                    for (int v = 0; v < 8; ++v)
                        a.u[v] = *(const u32*)&hls[arowbase + k0 + a_koff(v, hi)];
#pragma unroll
                    for (int gi = 0; gi < 4; ++gi) {
                        ABf bb;
#pragma unroll
                        for (int v = 0; v < 8; ++v)
                            bb.u[v] = bptr[gi][(k0 + a_koff(v, hi)) >> 1];
                        g[gi] = wmma_bf16(a, bb, g[gi]);
                    }
                }
            }
#pragma unroll
            for (int r = 0; r < 8; ++r) {
                float iv = sigf(g[0][r]);
                float fv = sigf(g[1][r]);
                float gv = tanhf(g[2][r]);
                float ov = sigf(g[3][r]);
                float cn = fv * c[mt][r] + iv * gv;
                float hn = ov * tanhf(cn);
                int brow = mt * 16 + r + hi * 8;
                bool active = (it == 0) || (dec_len[brow] > (it - 1));
                if (active) c[mt][r] = cn;
                hnew[mt][r] = hn;
                if (it > 0)               // raw h2 for preds (masking done in fc kernel)
                    H_all[((size_t)(it - 1) * B_ + brow) * H_ + ng * 16 + l15] = f2bf(hn);
            }
        }
        __syncthreads();                  // everyone done reading old h
#pragma unroll
        for (int mt = 0; mt < 2; ++mt)
#pragma unroll
            for (int r = 0; r < 8; ++r) {
                int brow = mt * 16 + r + hi * 8;
                bool active = (it == 0) || (dec_len[brow] > (it - 1));
                if (active) hls[brow * H_ + ng * 16 + l15] = f2bf(hnew[mt][r]);
            }
        wait_async0();                    // XW[it+1] staged before next iteration reads it
        __syncthreads();
    }
}

// ------------- preds = H_all @ fc_w.T + fc_b, masked by (dec_len > t), to d_out
__global__ void fc_kernel(const ushort_t* __restrict__ H_all, const ushort_t* __restrict__ fc_w_b,
                          const float* __restrict__ fc_b, const int* __restrict__ dec_len,
                          float* __restrict__ out) {
    const int MT = (T_ * B_) / 16; // 102
    const int NT = V_ / 16;        // 625
    int wid = (blockIdx.x * blockDim.x + threadIdx.x) >> 5;
    if (wid >= MT * NT) return;
    int lane = threadIdx.x & 31, hi = lane >> 4, l15 = lane & 15;
    int nt = wid % NT, mt = wid / NT;
    const u32* arow = (const u32*)(H_all + (size_t)(mt * 16 + l15) * H_);
    const u32* brow = (const u32*)(fc_w_b + (size_t)(nt * 16 + l15) * H_);
    f32x8 acc = {};
    for (int k0 = 0; k0 < H_; k0 += 32) {
        ABf a, bb;
#pragma unroll
        for (int v = 0; v < 8; ++v) {
            a.u[v]  = arow[(k0 + a_koff(v, hi)) >> 1];
            bb.u[v] = brow[(k0 + a_koff(v, hi)) >> 1];
        }
        acc = wmma_bf16(a, bb, acc);
    }
    int n = nt * 16 + l15;
    float bias = fc_b[n];
#pragma unroll
    for (int r = 0; r < 8; ++r) {
        int mrow = mt * 16 + r + hi * 8;
        int t = mrow >> 5, b = mrow & 31;
        bool active = dec_len[b] > t;
        out[(size_t)b * T_ * V_ + (size_t)t * V_ + n] = active ? (acc[r] + bias) : 0.f;
    }
}

// ---------------------------------------------------------------------------
extern "C" void kernel_launch(void* const* d_in, const int* in_sizes, int n_in,
                              void* d_out, int out_size, void* d_ws, size_t ws_size,
                              hipStream_t stream) {
    const float* enc      = (const float*)d_in[0];
    const int*   caps_in  = (const int*)  d_in[1];
    const int*   cap_len  = (const int*)  d_in[2];
    const float* emb_w    = (const float*)d_in[3];
    const float* init_w   = (const float*)d_in[4];
    const float* init_b   = (const float*)d_in[5];
    const float* bn_gamma = (const float*)d_in[6];
    const float* bn_beta  = (const float*)d_in[7];
    const float* w_ih     = (const float*)d_in[8];
    const float* w_hh     = (const float*)d_in[9];
    const float* b_ih     = (const float*)d_in[10];
    const float* b_hh     = (const float*)d_in[11];
    const float* fc_w     = (const float*)d_in[12];
    const float* fc_b     = (const float*)d_in[13];
    float* out = (float*)d_out;
    float* out_tail = out + (size_t)B_ * T_ * V_;

    char* ws = (char*)d_ws;
    size_t off = 0;
    auto alloc = [&](size_t bytes) -> void* {
        void* p = ws + off;
        off = (off + bytes + 255) & ~(size_t)255;
        return p;
    };
    int*      sort_ind    = (int*)alloc(B_ * 4);
    int*      dec_len     = (int*)alloc(B_ * 4);
    int*      caps_sorted = (int*)alloc(B_ * L_ * 4);
    float*    x0          = (float*)alloc((size_t)B_ * H_ * 4);
    ushort_t* x0bn        = (ushort_t*)alloc((size_t)B_ * H_ * 2);
    float*    b_comb      = (float*)alloc(G4_ * 4);
    ushort_t* w_ih_b      = (ushort_t*)alloc((size_t)G4_ * H_ * 2);
    ushort_t* w_hh_b      = (ushort_t*)alloc((size_t)G4_ * H_ * 2);
    ushort_t* fc_w_b      = (ushort_t*)alloc((size_t)V_ * H_ * 2);
    ushort_t* XW_b        = (ushort_t*)alloc((size_t)L_ * B_ * G4_ * 2);
    ushort_t* H_all       = (ushort_t*)alloc((size_t)T_ * B_ * H_ * 2);

    prep_kernel<<<1, 256, 0, stream>>>(caps_in, cap_len, init_b,
                                       sort_ind, dec_len, caps_sorted, x0, out_tail);
    convert_kernel<<<512, 256, 0, stream>>>(w_ih, w_hh, fc_w, b_ih, b_hh,
                                            w_ih_b, w_hh_b, fc_w_b, b_comb);
    {   // 32 n-tiles * 98 k-segments waves, 8 waves/block
        int waves = (H_ / 16) * 98;
        initmm_kernel<<<(waves + 7) / 8, 256, 0, stream>>>(enc, init_w, sort_ind, x0);
    }
    bn_kernel<<<1, H_, 0, stream>>>(x0, bn_gamma, bn_beta, x0bn);
    {
        int waves = ((L_ * B_) / 16) * (G4_ / 16);
        xw_kernel<<<(waves + 7) / 8, 256, 0, stream>>>(emb_w, caps_sorted, x0bn,
                                                       w_ih_b, b_comb, XW_b);
    }
    recur_kernel<<<1, 1024, 0, stream>>>(XW_b, w_hh_b, dec_len, H_all);
    {
        int waves = ((T_ * B_) / 16) * (V_ / 16);
        fc_kernel<<<(waves + 7) / 8, 256, 0, stream>>>(H_all, fc_w_b, fc_b, dec_len, out);
    }
}
